// VisionModel_72541997629650
// MI455X (gfx1250) — matrix operand order repaired and checked by reading, not compiled, not run on hardware
//
#include <hip/hip_runtime.h>
#include <hip/hip_bf16.h>

typedef __bf16 bf16;
typedef __attribute__((ext_vector_type(16))) bf16  v16bf;
typedef __attribute__((ext_vector_type(8)))  bf16  v8bf;
typedef __attribute__((ext_vector_type(8)))  float v8f;

#define BM 64
#define BN 128
#define BK 64
#define LDP (BK + 8)   // padded LDS row stride: 72 elems = 144B (16B aligned)

// ---------------------------------------------------------------------------
// Fragment helpers
// ---------------------------------------------------------------------------
__device__ __forceinline__ v16bf ld_frag16(const bf16* p0, const bf16* p1)
{
    v8bf lo = *(const v8bf*)p0;
    v8bf hi = *(const v8bf*)p1;
    return __builtin_shufflevector(lo, hi,
        0,1,2,3,4,5,6,7,8,9,10,11,12,13,14,15);
}

// One K-substep (K=32) at column offset koff within the staged BK=64 tile.
// Straight-line, no arrays: accumulators are independent references.
__device__ __forceinline__ void wmma_step(
    const bf16 (*As)[LDP], const bf16 (*Bs)[LDP], int koff,
    int wm, int wn, int half, int l15,
    v8f& c0, v8f& c1, v8f& c2, v8f& c3)
{
    const int ar = wm * 16 + l15;
    const int bc = wn * 64 + l15;
    v16bf a = ld_frag16(&As[ar][koff + half * 16], &As[ar][koff + half * 16 + 8]);

    v16bf b = ld_frag16(&Bs[bc][koff + half * 16], &Bs[bc][koff + half * 16 + 8]);
    c0 = __builtin_amdgcn_wmma_f32_16x16x32_bf16(false, a, false, b, (short)0, c0, false, false);
    b = ld_frag16(&Bs[bc + 16][koff + half * 16], &Bs[bc + 16][koff + half * 16 + 8]);
    c1 = __builtin_amdgcn_wmma_f32_16x16x32_bf16(false, a, false, b, (short)0, c1, false, false);
    b = ld_frag16(&Bs[bc + 32][koff + half * 16], &Bs[bc + 32][koff + half * 16 + 8]);
    c2 = __builtin_amdgcn_wmma_f32_16x16x32_bf16(false, a, false, b, (short)0, c2, false, false);
    b = ld_frag16(&Bs[bc + 48][koff + half * 16], &Bs[bc + 48][koff + half * 16 + 8]);
    c3 = __builtin_amdgcn_wmma_f32_16x16x32_bf16(false, a, false, b, (short)0, c3, false, false);
}

// ---------------------------------------------------------------------------
// Batched GEMM:  C[z](M,N) = A[z](M,K) * B[z](N,K)^T  (+bias)(+resid)(+act)
// Double-buffered LDS pipeline, ONE barrier per K-iteration:
//   issue loads(i+1) -> compute(i) -> ds_store(i+1 into buf^1) -> barrier
// All staging values are named scalars (no address-taken locals -> no spills).
// EXACT=true : requires M%64==0, N%128==0, K%64==0 -> zero guards
// EXACT=false: M/N via row clamping (OOB outputs masked at store),
//              K tail via one zero-filled tile
// act: 0 none, 1 quick-gelu x*sigmoid(1.702x), 2 exact gelu
// ---------------------------------------------------------------------------
template <bool EXACT>
__global__ __launch_bounds__(256, 1) void gemm_wmma_bf16(
    const bf16* __restrict__ A, int lda, long long sA,
    const bf16* __restrict__ B, int ldb, long long sB,
    float* __restrict__ C, bf16* __restrict__ Cb, int ldc, long long sC,
    const float* __restrict__ bias, const float* __restrict__ resid,
    int M, int N, int K, int act)
{
    __shared__ __align__(16) bf16 As[2][BM][LDP];
    __shared__ __align__(16) bf16 Bs[2][BN][LDP];

    const int tid  = threadIdx.x;
    const int lane = tid & 31;
    const int wave = tid >> 5;
    const int wm   = wave >> 1;      // 0..3 : 16 rows each
    const int wn   = wave & 1;       // 0..1 : 64 cols each
    const int half = lane >> 4;
    const int l15  = lane & 15;

    const int bm0 = blockIdx.y * BM;
    const int bn0 = blockIdx.x * BN;
    const long long z = blockIdx.z;

    A += z * sA;
    B += z * sB;

    // staging: 16B chunks (8 bf16); A tile 64x64 -> 2 chunks/thread,
    // B tile 128x64 -> 4 chunks/thread; chunk at (row, kk)
    const int rbase = tid >> 3;            // 0..31
    const int kk    = (tid & 7) << 3;      // 0,8,...,56
    const int ra1   = rbase + 32;

    const int gra0 = EXACT ? (bm0 + rbase) : min(bm0 + rbase, M - 1);
    const int gra1 = EXACT ? (bm0 + ra1)   : min(bm0 + ra1,   M - 1);
    const int gnb0 = EXACT ? (bn0 + rbase)      : min(bn0 + rbase,      N - 1);
    const int gnb1 = EXACT ? (bn0 + rbase + 32) : min(bn0 + rbase + 32, N - 1);
    const int gnb2 = EXACT ? (bn0 + rbase + 64) : min(bn0 + rbase + 64, N - 1);
    const int gnb3 = EXACT ? (bn0 + rbase + 96) : min(bn0 + rbase + 96, N - 1);

    const bf16* Ap0 = A + (long long)gra0 * lda + kk;
    const bf16* Ap1 = A + (long long)gra1 * lda + kk;
    const bf16* Bq0 = B + (long long)gnb0 * ldb + kk;
    const bf16* Bq1 = B + (long long)gnb1 * ldb + kk;
    const bf16* Bq2 = B + (long long)gnb2 * ldb + kk;
    const bf16* Bq3 = B + (long long)gnb3 * ldb + kk;

    v8f c0 = {}, c1 = {}, c2 = {}, c3 = {};

    const int kFull = EXACT ? K : (K & ~(BK - 1));
    const int nFull = kFull / BK;

    if (nFull > 0) {
        uint4 a0 = *(const uint4*)Ap0;
        uint4 a1 = *(const uint4*)Ap1;
        uint4 b0 = *(const uint4*)Bq0;
        uint4 b1 = *(const uint4*)Bq1;
        uint4 b2 = *(const uint4*)Bq2;
        uint4 b3 = *(const uint4*)Bq3;
        *(uint4*)&As[0][rbase][kk]      = a0;
        *(uint4*)&As[0][ra1][kk]        = a1;
        *(uint4*)&Bs[0][rbase][kk]      = b0;
        *(uint4*)&Bs[0][rbase + 32][kk] = b1;
        *(uint4*)&Bs[0][rbase + 64][kk] = b2;
        *(uint4*)&Bs[0][rbase + 96][kk] = b3;
    }
    __syncthreads();

    int cur = 0;
    for (int i = 0; i < nFull; ++i) {
        const bool more = (i + 1) < nFull;
        uint4 a0, a1, b0, b1, b2, b3;
        if (more) {
            const int kt = (i + 1) * BK;
            a0 = *(const uint4*)(Ap0 + kt);      // in flight during compute
            a1 = *(const uint4*)(Ap1 + kt);
            b0 = *(const uint4*)(Bq0 + kt);
            b1 = *(const uint4*)(Bq1 + kt);
            b2 = *(const uint4*)(Bq2 + kt);
            b3 = *(const uint4*)(Bq3 + kt);
            if (EXACT && (i + 2) < nFull) {
                __builtin_prefetch(Ap0 + (i + 2) * BK, 0, 1);
                __builtin_prefetch(Bq0 + (i + 2) * BK, 0, 1);
                __builtin_prefetch(Bq2 + (i + 2) * BK, 0, 1);
            }
        }
        wmma_step(As[cur], Bs[cur], 0,  wm, wn, half, l15, c0, c1, c2, c3);
        wmma_step(As[cur], Bs[cur], 32, wm, wn, half, l15, c0, c1, c2, c3);
        if (more) {
            const int nxt = cur ^ 1;      // no reader conflict: all read buf cur
            *(uint4*)&As[nxt][rbase][kk]      = a0;
            *(uint4*)&As[nxt][ra1][kk]        = a1;
            *(uint4*)&Bs[nxt][rbase][kk]      = b0;
            *(uint4*)&Bs[nxt][rbase + 32][kk] = b1;
            *(uint4*)&Bs[nxt][rbase + 64][kk] = b2;
            *(uint4*)&Bs[nxt][rbase + 96][kk] = b3;
            __syncthreads();              // publishes buf^1 AND frees buf cur
            cur = nxt;
        }
    }

    if (!EXACT && kFull < K) {
        __syncthreads();                  // buffer 'cur' free for reuse
        #pragma unroll
        for (int e = 0; e < 8; ++e) {
            int gk = kFull + kk + e;
            bool ok = gk < K;
            As[cur][rbase][kk + e]      = ok ? Ap0[kFull + e] : (bf16)0.0f;
            As[cur][ra1][kk + e]        = ok ? Ap1[kFull + e] : (bf16)0.0f;
            Bs[cur][rbase][kk + e]      = ok ? Bq0[kFull + e] : (bf16)0.0f;
            Bs[cur][rbase + 32][kk + e] = ok ? Bq1[kFull + e] : (bf16)0.0f;
            Bs[cur][rbase + 64][kk + e] = ok ? Bq2[kFull + e] : (bf16)0.0f;
            Bs[cur][rbase + 96][kk + e] = ok ? Bq3[kFull + e] : (bf16)0.0f;
        }
        __syncthreads();
        wmma_step(As[cur], Bs[cur], 0,  wm, wn, half, l15, c0, c1, c2, c3);
        wmma_step(As[cur], Bs[cur], 32, wm, wn, half, l15, c0, c1, c2, c3);
    }

    // ---- epilogue: C VGPR i holds M = half*8 + i, N = l15 ----
    v8f accs[4] = {c0, c1, c2, c3};
    #pragma unroll
    for (int t = 0; t < 4; ++t) {
        #pragma unroll
        for (int i = 0; i < 8; ++i) {
            int row = bm0 + wm * 16 + half * 8 + i;
            int col = bn0 + wn * 64 + t * 16 + l15;
            if (EXACT || (row < M && col < N)) {
                float v = accs[t][i];
                if (bias)  v += bias[col];
                long long cidx = z * sC + (long long)row * ldc + col;
                if (resid) v += resid[cidx];
                if (act == 1)       v = v / (1.0f + __expf(-1.702f * v));
                else if (act == 2)  v = 0.5f * v * (1.0f + erff(v * 0.70710678118f));
                if (C)  C[cidx]  = v;
                if (Cb) Cb[cidx] = (bf16)v;
            }
        }
    }
}

// ---------------------------------------------------------------------------
// fp32 -> bf16 cast (grid-stride)
// ---------------------------------------------------------------------------
__global__ void cast_f32_bf16(const float* __restrict__ in, bf16* __restrict__ out,
                              long long n)
{
    long long i = (long long)blockIdx.x * blockDim.x + threadIdx.x;
    long long stride = (long long)gridDim.x * blockDim.x;
    for (; i < n; i += stride) out[i] = (bf16)in[i];
}

// ---------------------------------------------------------------------------
// LayerNorm over last dim, one block per row, bf16 output
// ---------------------------------------------------------------------------
__global__ __launch_bounds__(256) void layernorm_bf16(
    const float* __restrict__ x, const float* __restrict__ w,
    const float* __restrict__ b, bf16* __restrict__ o, int Dd)
{
    const int tid = threadIdx.x;
    const long long row = blockIdx.x;
    const float* xr = x + row * Dd;
    bf16* orow = o + row * Dd;

    __shared__ float red[256];
    float s = 0.0f, ss = 0.0f;
    for (int i = tid; i < Dd; i += 256) { float v = xr[i]; s += v; ss += v * v; }

    red[tid] = s; __syncthreads();
    for (int off = 128; off > 0; off >>= 1) {
        if (tid < off) red[tid] += red[tid + off];
        __syncthreads();
    }
    float mean = red[0] / (float)Dd;
    __syncthreads();

    red[tid] = ss; __syncthreads();
    for (int off = 128; off > 0; off >>= 1) {
        if (tid < off) red[tid] += red[tid + off];
        __syncthreads();
    }
    float var = red[0] / (float)Dd - mean * mean;
    float rstd = rsqrtf(var + 1e-6f);

    for (int i = tid; i < Dd; i += 256)
        orow[i] = (bf16)((xr[i] - mean) * rstd * w[i] + b[i]);
}

// ---------------------------------------------------------------------------
// RoPE + split qkv(S,3,H,HD) fp32 -> q,k bf16 [H][S][HD], v bf16 transposed [H][HD][S]
// ---------------------------------------------------------------------------
__global__ __launch_bounds__(256) void rope_split(
    const float* __restrict__ qkv, const float* __restrict__ rot,
    bf16* __restrict__ qb, bf16* __restrict__ kb, bf16* __restrict__ vt,
    int S, int Hn, int HD)
{
    const int s  = blockIdx.x;
    const int DD = Hn * HD;
    const float* row = qkv + (long long)s * 3 * DD;
    const float* r40 = rot + (long long)s * 40;

    for (int j = threadIdx.x; j < DD; j += 256) {
        int h = j / HD;
        int d = j - h * HD;
        float c  = cosf(r40[d % 40]);
        float sn = sinf(r40[d % 40]);
        int jp = (d < 40) ? (j + 40) : (j - 40);
        float sgn = (d < 40) ? -1.0f : 1.0f;

        float q  = row[j];
        float k  = row[DD + j];
        float v  = row[2 * DD + j];
        float qr = q * c + sgn * row[jp] * sn;
        float kr = k * c + sgn * row[DD + jp] * sn;

        long long qi = (long long)h * S * HD + (long long)s * HD + d;
        qb[qi] = (bf16)qr;
        kb[qi] = (bf16)kr;
        vt[(long long)h * HD * S + (long long)d * S + s] = (bf16)v;
    }
}

// ---------------------------------------------------------------------------
// Masked softmax over rows of scores[H][S][S]; scale folded in; bf16 probs out
// ---------------------------------------------------------------------------
__global__ __launch_bounds__(256) void masked_softmax(
    const float* __restrict__ sc, const int* __restrict__ seg,
    bf16* __restrict__ p, int S, float scale)
{
    const int tid = threadIdx.x;
    const int s = blockIdx.x % S;
    const int h = blockIdx.x / S;
    const float* row = sc + ((long long)h * S + s) * S;
    bf16* prow = p + ((long long)h * S + s) * S;
    const int myseg = seg[s];

    __shared__ float red[256];

    float m = -3.4e38f;
    for (int t = tid; t < S; t += 256)
        if (seg[t] == myseg) m = fmaxf(m, row[t] * scale);
    red[tid] = m; __syncthreads();
    for (int off = 128; off > 0; off >>= 1) {
        if (tid < off) red[tid] = fmaxf(red[tid], red[tid + off]);
        __syncthreads();
    }
    m = red[0]; __syncthreads();

    float sum = 0.0f;
    for (int t = tid; t < S; t += 256)
        if (seg[t] == myseg) sum += __expf(row[t] * scale - m);
    red[tid] = sum; __syncthreads();
    for (int off = 128; off > 0; off >>= 1) {
        if (tid < off) red[tid] += red[tid + off];
        __syncthreads();
    }
    float inv = 1.0f / red[0];

    for (int t = tid; t < S; t += 256) {
        float pv = (seg[t] == myseg) ? __expf(row[t] * scale - m) * inv : 0.0f;
        prow[t] = (bf16)pv;
    }
}

// ---------------------------------------------------------------------------
// Host-side helpers
// ---------------------------------------------------------------------------
static void launch_gemm(hipStream_t st,
    const bf16* A, int lda, long long sA,
    const bf16* B, int ldb, long long sB,
    float* C, bf16* Cb, int ldc, long long sC,
    const float* bias, const float* resid,
    int M, int N, int K, int act, int batch)
{
    dim3 grid((N + BN - 1) / BN, (M + BM - 1) / BM, batch);
    bool exact = (M % BM == 0) && (N % BN == 0) && (K % BK == 0);
    if (exact)
        gemm_wmma_bf16<true><<<grid, 256, 0, st>>>(A, lda, sA, B, ldb, sB,
            C, Cb, ldc, sC, bias, resid, M, N, K, act);
    else
        gemm_wmma_bf16<false><<<grid, 256, 0, st>>>(A, lda, sA, B, ldb, sB,
            C, Cb, ldc, sC, bias, resid, M, N, K, act);
}

static void castN(hipStream_t st, const float* src, bf16* dst, long long n)
{
    long long blocks = (n + 255) / 256;
    if (blocks > 16384) blocks = 16384;
    cast_f32_bf16<<<(int)blocks, 256, 0, st>>>(src, dst, n);
}

extern "C" void kernel_launch(void* const* d_in, const int* in_sizes, int n_in,
                              void* d_out, int out_size, void* d_ws, size_t ws_size,
                              hipStream_t stream)
{
    (void)in_sizes; (void)n_in; (void)out_size; (void)ws_size;

    const int S = 1280, Dm = 1280, Hn = 16, HDm = 80, FF = 5120;
    const int KP = 1176, MR = 320, OD = 1536;
    const int DEPTH = 4;

    const float* pixels = (const float*)d_in[0];
    const float* rotary = (const float*)d_in[1];
    const int*   seg    = (const int*)d_in[2];
    const float* proj_w = (const float*)d_in[3];
    const float* ln1_w  = (const float*)d_in[4];
    const float* ln1_b  = (const float*)d_in[5];
    const float* qkv_w  = (const float*)d_in[6];
    const float* qkv_b  = (const float*)d_in[7];
    const float* po_w   = (const float*)d_in[8];
    const float* po_b   = (const float*)d_in[9];
    const float* ln2_w  = (const float*)d_in[10];
    const float* ln2_b  = (const float*)d_in[11];
    const float* fc1_w  = (const float*)d_in[12];
    const float* fc1_b  = (const float*)d_in[13];
    const float* fc2_w  = (const float*)d_in[14];
    const float* fc2_b  = (const float*)d_in[15];
    const float* mln_w  = (const float*)d_in[16];
    const float* mln_b  = (const float*)d_in[17];
    const float* m1_w   = (const float*)d_in[18];
    const float* m1_b   = (const float*)d_in[19];
    const float* m2_w   = (const float*)d_in[20];
    const float* m2_b   = (const float*)d_in[21];
    float* out = (float*)d_out;

    // ---- bump allocator over workspace ----
    char* base = (char*)d_ws;
    size_t off = 0;
    auto alloc = [&](size_t bytes) -> void* {
        void* p = base + off;
        off = (off + bytes + 255) & ~(size_t)255;
        return p;
    };

    bf16*  pix_bf = (bf16*) alloc((size_t)S * KP * 2);
    float* x      = (float*)alloc((size_t)S * Dm * 4);
    bf16*  h_bf   = (bf16*) alloc((size_t)S * Dm * 2);
    float* qkv_f  = (float*)alloc((size_t)S * 3 * Dm * 4);
    bf16*  q_bf   = (bf16*) alloc((size_t)S * Dm * 2);
    bf16*  k_bf   = (bf16*) alloc((size_t)S * Dm * 2);
    bf16*  v_t    = (bf16*) alloc((size_t)S * Dm * 2);
    float* scores = (float*)alloc((size_t)Hn * S * S * 4);
    bf16*  p_bf   = (bf16*) alloc((size_t)Hn * S * S * 2);
    bf16*  o_bf   = (bf16*) alloc((size_t)S * Dm * 2);
    bf16*  t_bf   = (bf16*) alloc((size_t)S * FF * 2);
    bf16*  wb     = (bf16*) alloc((size_t)FF * FF * 2);   // reusable weight slab

    const float scale = 0.11180339887f;  // 1/sqrt(80)

    // ---- patch projection: x = pixels @ proj_w^T ----
    castN(stream, pixels, pix_bf, (long long)S * KP);
    castN(stream, proj_w, wb, (long long)Dm * KP);
    launch_gemm(stream, pix_bf, KP, 0, wb, KP, 0,
                x, nullptr, Dm, 0, nullptr, nullptr, S, Dm, KP, 0, 1);

    for (int l = 0; l < DEPTH; ++l) {
        // h = LN1(x)
        layernorm_bf16<<<S, 256, 0, stream>>>(x, ln1_w + l * Dm, ln1_b + l * Dm, h_bf, Dm);

        // qkv = h @ qkv_w^T + qkv_b
        castN(stream, qkv_w + (long long)l * 3 * Dm * Dm, wb, (long long)3 * Dm * Dm);
        launch_gemm(stream, h_bf, Dm, 0, wb, Dm, 0,
                    qkv_f, nullptr, 3 * Dm, 0, qkv_b + l * 3 * Dm, nullptr,
                    S, 3 * Dm, Dm, 0, 1);

        // rope + split
        rope_split<<<S, 256, 0, stream>>>(qkv_f, rotary, q_bf, k_bf, v_t, S, Hn, HDm);

        // scores[h] = q_h @ k_h^T   (batched over heads)
        launch_gemm(stream, q_bf, HDm, (long long)S * HDm,
                    k_bf, HDm, (long long)S * HDm,
                    scores, nullptr, S, (long long)S * S, nullptr, nullptr,
                    S, S, HDm, 0, Hn);

        // masked softmax (scale folded in), bf16 probs
        masked_softmax<<<Hn * S, 256, 0, stream>>>(scores, seg, p_bf, S, scale);

        // o[:, h*80:(h+1)*80] = p_h @ v_h   (v stored transposed: [H][HD][S])
        launch_gemm(stream, p_bf, S, (long long)S * S,
                    v_t, S, (long long)HDm * S,
                    nullptr, o_bf, Dm, (long long)HDm, nullptr, nullptr,
                    S, HDm, S, 0, Hn);

        // x = x + o @ po_w^T + po_b
        castN(stream, po_w + (long long)l * Dm * Dm, wb, (long long)Dm * Dm);
        launch_gemm(stream, o_bf, Dm, 0, wb, Dm, 0,
                    x, nullptr, Dm, 0, po_b + l * Dm, x, S, Dm, Dm, 0, 1);

        // h = LN2(x)
        layernorm_bf16<<<S, 256, 0, stream>>>(x, ln2_w + l * Dm, ln2_b + l * Dm, h_bf, Dm);

        // t = quickgelu(h @ fc1_w^T + fc1_b)  (bf16 out)
        castN(stream, fc1_w + (long long)l * FF * Dm, wb, (long long)FF * Dm);
        launch_gemm(stream, h_bf, Dm, 0, wb, Dm, 0,
                    nullptr, t_bf, FF, 0, fc1_b + l * FF, nullptr,
                    S, FF, Dm, 1, 1);

        // x = x + t @ fc2_w^T + fc2_b
        castN(stream, fc2_w + (long long)l * Dm * FF, wb, (long long)Dm * FF);
        launch_gemm(stream, t_bf, FF, 0, wb, FF, 0,
                    x, nullptr, Dm, 0, fc2_b + l * Dm, x, S, Dm, FF, 0, 1);
    }

    // ---- merge head: view LN output as (320, 5120) ----
    layernorm_bf16<<<S, 256, 0, stream>>>(x, mln_w, mln_b, h_bf, Dm);

    castN(stream, m1_w, wb, (long long)FF * FF);
    launch_gemm(stream, h_bf, FF, 0, wb, FF, 0,
                nullptr, t_bf, FF, 0, m1_b, nullptr, MR, FF, FF, 2, 1);

    castN(stream, m2_w, wb, (long long)OD * FF);
    launch_gemm(stream, t_bf, FF, 0, wb, FF, 0,
                out, nullptr, OD, 0, m2_b, nullptr, MR, OD, FF, 0, 1);
}
